// CrossModalAttention_88424786690605
// MI455X (gfx1250) — matrix operand order repaired and checked by reading, not compiled, not run on hardware
//
#include <hip/hip_runtime.h>

#define S_LEN 1024
#define NHEAD 16

typedef __attribute__((ext_vector_type(16))) __bf16 v16bf;
typedef __attribute__((ext_vector_type(8)))  float  v8f;

union BFrag { v16bf v; uint4 q[2]; };

// A-matrix 16x32 bf16 fragment: lane half 0 holds K {0..7,16..23}, half 1 holds {8..15,24..31}
__device__ __forceinline__ void load_a_bf16(BFrag& f, const __bf16* p0) {
    f.q[0] = *(const uint4*)(p0);        // 8 bf16 at k0 + half*8
    f.q[1] = *(const uint4*)(p0 + 16);   // 8 bf16 at k0 + half*8 + 16
}
__device__ __forceinline__ void load_a_f32cvt(BFrag& f, const float* p0) {
#pragma unroll
    for (int i = 0; i < 8; ++i) { f.v[i] = (__bf16)p0[i]; f.v[8 + i] = (__bf16)p0[16 + i]; }
}
// B-matrix 32x16 bf16 fragment: lane half 0 holds K {0..7,16..23}, half 1 holds {8..15,24..31}
__device__ __forceinline__ void load_b_bf16(BFrag& f, const __bf16* p0) {
    f.q[0] = *(const uint4*)(p0);
    f.q[1] = *(const uint4*)(p0 + 16);
}

// Unified NT GEMM: C[m,n] = sum_k A[m,k] * Bt[n,k]   (Bt row-major [N,K])
// Each wave computes a 32(M) x 64(N) strip (2 M-tiles reuse 4 B fragments);
// 4 waves/block -> 128x64 tile per block.
// MODE: 0 f32(+bias)+opt bf16 | 1 QK per-head bf16 [B,H,S,64] | 2 V^T bf16 [B,H,64,S]
//       3 f32 = acc+bias+res  | 5 sigmoid-gate fuse -> bf16   | 6 scaled f32 (batched scores)
//       7 ctx bf16 [B,S,H*64] (batched)
template <int MODE, int AF32>
__global__ __launch_bounds__(128)
void wmma_gemm(const void* __restrict__ Ap, long long aBatch, int lda,
               const __bf16* __restrict__ Bp, long long bBatch, int ldb,
               float* __restrict__ Cf, __bf16* __restrict__ Cb, long long cBatch, int ldc,
               const float* __restrict__ bias,
               const float* __restrict__ aux0, const float* __restrict__ aux1,
               const float* __restrict__ aux2,
               int K, float scale)
{
    const int wave  = threadIdx.x >> 5;
    const int lane  = threadIdx.x & 31;
    const int half  = lane >> 4;
    const int r     = lane & 15;
    const int z     = blockIdx.z;
    const int mBase = blockIdx.y * 128 + wave * 32;   // wave strip: rows [mBase, mBase+32)
    const int nBase = blockIdx.x * 64;

    // per-M-tile A row pointers
    const __bf16* Abf[2] = {nullptr, nullptr};
    const float*  Af[2]  = {nullptr, nullptr};
#pragma unroll
    for (int t = 0; t < 2; ++t) {
        const int row = mBase + t * 16 + r;
        if (AF32) Af[t]  = (const float*)Ap  + (size_t)z * aBatch + (size_t)row * lda + half * 8;
        else      Abf[t] = (const __bf16*)Ap + (size_t)z * aBatch + (size_t)row * lda + half * 8;
    }

    // per-j B row pointers (shared by both M tiles)
    const __bf16* Bq[4];
#pragma unroll
    for (int j = 0; j < 4; ++j)
        Bq[j] = Bp + (size_t)z * bBatch + (size_t)(nBase + j * 16 + r) * ldb + half * 8;

    v8f acc[2][4];
#pragma unroll
    for (int t = 0; t < 2; ++t)
#pragma unroll
        for (int j = 0; j < 4; ++j)
#pragma unroll
            for (int i = 0; i < 8; ++i) acc[t][j][i] = 0.f;

#pragma unroll 2
    for (int k0 = 0; k0 < K; k0 += 32) {
        BFrag a[2], bf[4];
#pragma unroll
        for (int t = 0; t < 2; ++t) {
            if (AF32) load_a_f32cvt(a[t], Af[t] + k0);
            else      load_a_bf16(a[t], Abf[t] + k0);
        }
#pragma unroll
        for (int j = 0; j < 4; ++j)
            load_b_bf16(bf[j], Bq[j] + k0);
        if (AF32) __builtin_prefetch((const void*)(Af[0] + k0 + 64), 0, 3);
        else      __builtin_prefetch((const void*)(Abf[0] + k0 + 64), 0, 3);
#pragma unroll
        for (int t = 0; t < 2; ++t)
#pragma unroll
            for (int j = 0; j < 4; ++j)
                acc[t][j] = __builtin_amdgcn_wmma_f32_16x16x32_bf16(
                    false, a[t].v, false, bf[j].v, (short)0, acc[t][j], false, false);
    }

#pragma unroll
    for (int t = 0; t < 2; ++t) {
        const int mTile = mBase + t * 16;
#pragma unroll
        for (int j = 0; j < 4; ++j) {
            const int n  = nBase + j * 16 + r;
            const float bv = (MODE == 0 || MODE == 1 || MODE == 2 || MODE == 3) ? bias[n] : 0.f;
#pragma unroll
            for (int i = 0; i < 8; ++i) {
                const int m = mTile + i + half * 8;
                float v = acc[t][j][i];
                if (MODE == 0) {
                    v += bv;
                    size_t idx = (size_t)m * ldc + n;
                    Cf[idx] = v;
                    if (Cb) Cb[idx] = (__bf16)v;
                } else if (MODE == 1) {
                    v += bv;
                    int b = m >> 10, s = m & (S_LEN - 1);
                    int h = n >> 6,  d = n & 63;
                    Cb[(((size_t)(b * NHEAD + h) * S_LEN + s) << 6) + d] = (__bf16)v;
                } else if (MODE == 2) {
                    v += bv;
                    int b = m >> 10, s = m & (S_LEN - 1);
                    int h = n >> 6,  d = n & 63;
                    Cb[((size_t)(b * NHEAD + h) * 64 + d) * S_LEN + s] = (__bf16)v;
                } else if (MODE == 3) {
                    size_t idx = (size_t)m * ldc + n;
                    Cf[idx] = v + bv + aux0[idx];
                } else if (MODE == 5) {
                    size_t idx = (size_t)m * ldc + n;
                    float g = 1.f / (1.f + __expf(-(aux0[idx] + v)));
                    Cb[idx] = (__bf16)(g * aux1[idx] + (1.f - g) * aux2[idx]);
                } else if (MODE == 6) {
                    Cf[(size_t)z * cBatch + (size_t)m * ldc + n] = v * scale;
                } else { // 7: ctx
                    int b = z >> 4, h = z & 15;
                    Cb[((size_t)(b * S_LEN + m) * 1024) + h * 64 + n] = (__bf16)v;
                }
            }
        }
    }
}

__global__ __launch_bounds__(256)
void transpose_f32_to_bf16(const float* __restrict__ W, __bf16* __restrict__ Wt,
                           int Kd, int Nd)
{
    size_t i = (size_t)blockIdx.x * 256 + threadIdx.x;
    if (i >= (size_t)Kd * Nd) return;
    int k = (int)(i / Nd), n = (int)(i % Nd);
    Wt[(size_t)n * Kd + k] = (__bf16)W[i];
}

__global__ __launch_bounds__(256)
void softmax_rows(float* __restrict__ data, int n) // n == 1024
{
    __shared__ float red[8];
    float* row = data + (size_t)blockIdx.x * n;
    int t = threadIdx.x;
    float4 x = ((float4*)row)[t];
    float m = fmaxf(fmaxf(x.x, x.y), fmaxf(x.z, x.w));
    for (int o = 16; o > 0; o >>= 1) m = fmaxf(m, __shfl_xor(m, o, 32));
    if ((t & 31) == 0) red[t >> 5] = m;
    __syncthreads();
    float M = red[0];
#pragma unroll
    for (int w = 1; w < 8; ++w) M = fmaxf(M, red[w]);
    __syncthreads();
    x.x = __expf(x.x - M); x.y = __expf(x.y - M);
    x.z = __expf(x.z - M); x.w = __expf(x.w - M);
    float s = x.x + x.y + x.z + x.w;
    for (int o = 16; o > 0; o >>= 1) s += __shfl_xor(s, o, 32);
    if ((t & 31) == 0) red[t >> 5] = s;
    __syncthreads();
    float S = 0.f;
#pragma unroll
    for (int w = 0; w < 8; ++w) S += red[w];
    float inv = 1.f / S;
    x.x *= inv; x.y *= inv; x.z *= inv; x.w *= inv;
    ((float4*)row)[t] = x;
}

__global__ __launch_bounds__(256)
void layernorm_rows(const float* __restrict__ X, const float* __restrict__ g,
                    const float* __restrict__ bta, float* __restrict__ Yf,
                    __bf16* __restrict__ Yb, int N)
{
    __shared__ float s1[8], s2[8];
    const float* x = X + (size_t)blockIdx.x * N;
    int t = threadIdx.x;
    float s = 0.f, q = 0.f;
    for (int i = t; i < N; i += 256) { float v = x[i]; s += v; q += v * v; }
    for (int o = 16; o > 0; o >>= 1) { s += __shfl_xor(s, o, 32); q += __shfl_xor(q, o, 32); }
    if ((t & 31) == 0) { s1[t >> 5] = s; s2[t >> 5] = q; }
    __syncthreads();
    float S = 0.f, Q = 0.f;
#pragma unroll
    for (int w = 0; w < 8; ++w) { S += s1[w]; Q += s2[w]; }
    float mu = S / N;
    float var = Q / N - mu * mu;
    float rs = rsqrtf(var + 1e-5f);
    float* yf = Yf + (size_t)blockIdx.x * N;
    __bf16* yb = Yb ? (Yb + (size_t)blockIdx.x * N) : nullptr;
    for (int i = t; i < N; i += 256) {
        float v = (x[i] - mu) * rs * g[i] + bta[i];
        yf[i] = v;
        if (yb) yb[i] = (__bf16)v;
    }
}

extern "C" void kernel_launch(void* const* d_in, const int* in_sizes, int n_in,
                              void* d_out, int out_size, void* d_ws, size_t ws_size,
                              hipStream_t stream) {
    (void)in_sizes; (void)n_in; (void)out_size; (void)ws_size;
    auto in = [&](int i) { return (const float*)d_in[i]; };

    // ---- workspace layout ----
    char* ws = (char*)d_ws;
    size_t off = 0;
    auto takeB = [&](size_t e) { __bf16* p = (__bf16*)(ws + off); off = (off + e * 2 + 255) & ~(size_t)255; return p; };
    auto takeF = [&](size_t e) { float*  p = (float*)(ws + off);  off = (off + e * 4 + 255) & ~(size_t)255; return p; };

    __bf16* wt_litproj = takeB((size_t)768 * 1024);
    __bf16* wt_kgproj  = takeB((size_t)512 * 1024);
    __bf16 *wt_q[2], *wt_k[2], *wt_v[2], *wt_o[2];
    for (int i = 0; i < 2; ++i) {
        wt_q[i] = takeB((size_t)1024 * 1024); wt_k[i] = takeB((size_t)1024 * 1024);
        wt_v[i] = takeB((size_t)1024 * 1024); wt_o[i] = takeB((size_t)1024 * 1024);
    }
    __bf16* wt_litgate = takeB((size_t)2048 * 1024);
    __bf16* wt_kggate  = takeB((size_t)2048 * 1024);
    __bf16* wt_litout  = takeB((size_t)1024 * 768);
    __bf16* wt_kgout   = takeB((size_t)1024 * 512);

    const size_t MH = (size_t)4096 * 1024;
    float*  litP_f  = takeF(MH);  float*  kgP_f   = takeF(MH);
    __bf16* litP_b  = takeB(MH);  __bf16* kgP_b   = takeB(MH);
    __bf16* Qh      = takeB(MH);  __bf16* Kh      = takeB(MH);
    __bf16* Vt      = takeB(MH);  __bf16* ctx_b   = takeB(MH);
    float*  attnO_f = takeF(MH);
    float*  litAtt_f = takeF(MH); float*  kgAtt_f = takeF(MH);
    __bf16* litAtt_b = takeB(MH); __bf16* kgAtt_b = takeB(MH);
    float*  gate_f  = takeF(MH);
    __bf16* fused_b = takeB(MH);
    float*  pre_f   = takeF((size_t)4096 * 768);

    // ---- output regions ----
    float* out     = (float*)d_out;
    float* out_lit = out;
    float* out_kg  = out + (size_t)4096 * 768;
    float* attn[2];
    attn[0] = out_kg + (size_t)4096 * 512;
    attn[1] = attn[0] + (size_t)64 * 1024 * 1024;

    auto tr = [&](const float* W, __bf16* Wt, int Kd, int Nd) {
        size_t tot = (size_t)Kd * Nd;
        transpose_f32_to_bf16<<<dim3((unsigned)((tot + 255) / 256)), dim3(256), 0, stream>>>(W, Wt, Kd, Nd);
    };
    tr(in(2),  wt_litproj, 768, 1024);
    tr(in(4),  wt_kgproj,  512, 1024);
    tr(in(6),  wt_q[0], 1024, 1024); tr(in(8),  wt_k[0], 1024, 1024);
    tr(in(10), wt_v[0], 1024, 1024); tr(in(12), wt_o[0], 1024, 1024);
    tr(in(16), wt_q[1], 1024, 1024); tr(in(18), wt_k[1], 1024, 1024);
    tr(in(20), wt_v[1], 1024, 1024); tr(in(22), wt_o[1], 1024, 1024);
    tr(in(26), wt_litgate, 2048, 1024);
    tr(in(28), wt_kggate,  2048, 1024);
    tr(in(30), wt_litout,  1024, 768);
    tr(in(32), wt_kgout,   1024, 512);

    auto gemm = [&](const void* A, int aF32, long long aB, int lda,
                    const __bf16* B, long long bB, int ldb,
                    float* Cf, __bf16* Cb, long long cB, int ldc,
                    const float* bias, const float* a0, const float* a1, const float* a2,
                    int M, int N, int K, int mode, float scale, int batches) {
        dim3 g((unsigned)(N / 64), (unsigned)(M / 128), (unsigned)batches);
#define GEMM_ARGS A, aB, lda, B, bB, ldb, Cf, Cb, cB, ldc, bias, a0, a1, a2, K, scale
        int sel = mode * 2 + aF32;
        switch (sel) {
            case 0*2+0: wmma_gemm<0,0><<<g, dim3(128), 0, stream>>>(GEMM_ARGS); break;
            case 0*2+1: wmma_gemm<0,1><<<g, dim3(128), 0, stream>>>(GEMM_ARGS); break;
            case 1*2+0: wmma_gemm<1,0><<<g, dim3(128), 0, stream>>>(GEMM_ARGS); break;
            case 2*2+0: wmma_gemm<2,0><<<g, dim3(128), 0, stream>>>(GEMM_ARGS); break;
            case 3*2+0: wmma_gemm<3,0><<<g, dim3(128), 0, stream>>>(GEMM_ARGS); break;
            case 5*2+0: wmma_gemm<5,0><<<g, dim3(128), 0, stream>>>(GEMM_ARGS); break;
            case 6*2+0: wmma_gemm<6,0><<<g, dim3(128), 0, stream>>>(GEMM_ARGS); break;
            case 7*2+1: wmma_gemm<7,1><<<g, dim3(128), 0, stream>>>(GEMM_ARGS); break;
            default: break;
        }
#undef GEMM_ARGS
    };

    // input projections (A read as f32, converted on load)
    gemm(in(0), 1, 0, 768, wt_litproj, 0, 768, litP_f, litP_b, 0, 1024,
         in(3), nullptr, nullptr, nullptr, 4096, 1024, 768, 0, 1.f, 1);
    gemm(in(1), 1, 0, 512, wt_kgproj, 0, 512, kgP_f, kgP_b, 0, 1024,
         in(5), nullptr, nullptr, nullptr, 4096, 1024, 512, 0, 1.f, 1);

    // two cross-attentions: s2==0 -> l2k (q=lit, kv=kg), s2==1 -> k2l
    for (int s2 = 0; s2 < 2; ++s2) {
        const __bf16* q_src  = (s2 == 0) ? litP_b : kgP_b;
        const __bf16* kv_src = (s2 == 0) ? kgP_b  : litP_b;
        const float*  proj_f = (s2 == 0) ? litP_f : kgP_f;
        int base = (s2 == 0) ? 6 : 16;
        gemm(q_src, 0, 0, 1024, wt_q[s2], 0, 1024, nullptr, Qh, 0, 1024,
             in(base + 1), nullptr, nullptr, nullptr, 4096, 1024, 1024, 1, 1.f, 1);
        gemm(kv_src, 0, 0, 1024, wt_k[s2], 0, 1024, nullptr, Kh, 0, 1024,
             in(base + 3), nullptr, nullptr, nullptr, 4096, 1024, 1024, 1, 1.f, 1);
        gemm(kv_src, 0, 0, 1024, wt_v[s2], 0, 1024, nullptr, Vt, 0, 1024,
             in(base + 5), nullptr, nullptr, nullptr, 4096, 1024, 1024, 2, 1.f, 1);
        // scores: per (b,h) 1024x1024, K=64, scale 1/sqrt(64)
        gemm(Qh, 0, (long long)1024 * 64, 64, Kh, (long long)1024 * 64, 64,
             attn[s2], nullptr, (long long)1024 * 1024, 1024,
             nullptr, nullptr, nullptr, nullptr, 1024, 1024, 64, 6, 0.125f, 64);
        softmax_rows<<<dim3(65536), dim3(256), 0, stream>>>(attn[s2], 1024);
        // ctx = attn(f32->bf16 on load) @ V^T-rows
        gemm(attn[s2], 1, (long long)1024 * 1024, 1024, Vt, (long long)64 * 1024, 1024,
             nullptr, ctx_b, 0, 0, nullptr, nullptr, nullptr, nullptr,
             1024, 64, 1024, 7, 1.f, 64);
        // wo + residual(q) -> pre-LN
        gemm(ctx_b, 0, 0, 1024, wt_o[s2], 0, 1024, attnO_f, nullptr, 0, 1024,
             in(base + 7), proj_f, nullptr, nullptr, 4096, 1024, 1024, 3, 1.f, 1);
        float*  att_f = (s2 == 0) ? litAtt_f : kgAtt_f;
        __bf16* att_b = (s2 == 0) ? litAtt_b : kgAtt_b;
        layernorm_rows<<<dim3(4096), dim3(256), 0, stream>>>(
            attnO_f, in(base + 8), in(base + 9), att_f, att_b, 1024);
    }

    // gate fusion + output projection + final LN
    for (int s2 = 0; s2 < 2; ++s2) {
        const __bf16* proj_b = (s2 == 0) ? litP_b : kgP_b;
        const float*  proj_f = (s2 == 0) ? litP_f : kgP_f;
        const __bf16* att_b  = (s2 == 0) ? litAtt_b : kgAtt_b;
        const float*  att_f  = (s2 == 0) ? litAtt_f : kgAtt_f;
        const __bf16* wtg    = (s2 == 0) ? wt_litgate : wt_kggate;
        const float*  bg     = (s2 == 0) ? in(27) : in(29);
        const __bf16* wto2   = (s2 == 0) ? wt_litout : wt_kgout;
        const float*  bo2    = (s2 == 0) ? in(31) : in(33);
        int OUTD             = (s2 == 0) ? 768 : 512;
        const float*  resid  = (s2 == 0) ? in(0) : in(1);
        const float*  lng    = (s2 == 0) ? in(34) : in(36);
        const float*  lnb    = (s2 == 0) ? in(35) : in(37);
        float*        dst    = (s2 == 0) ? out_lit : out_kg;
        // gate pass1: proj @ W_top + bias
        gemm(proj_b, 0, 0, 1024, wtg, 0, 2048, gate_f, nullptr, 0, 1024,
             bg, nullptr, nullptr, nullptr, 4096, 1024, 1024, 0, 1.f, 1);
        // gate pass2: + att @ W_bot, sigmoid, fuse -> bf16
        gemm(att_b, 0, 0, 1024, wtg + 1024, 0, 2048, nullptr, fused_b, 0, 1024,
             nullptr, gate_f, proj_f, att_f, 4096, 1024, 1024, 5, 1.f, 1);
        // out projection + residual(features)
        gemm(fused_b, 0, 0, 1024, wto2, 0, 1024, pre_f, nullptr, 0, OUTD,
             bo2, resid, nullptr, nullptr, 4096, OUTD, 1024, 3, 1.f, 1);
        layernorm_rows<<<dim3(4096), dim3(256), 0, stream>>>(
            pre_f, lng, lnb, dst, nullptr, OUTD);
    }
}
